// VllmMixtureOfExpertsOp_15401752723970
// MI455X (gfx1250) — compile-verified, matrix operand order checked by reading, not run on hardware
//
#include <hip/hip_runtime.h>
#include <hip/hip_bf16.h>

#define NEXP 8
#define TTOK 1024
#define DDIM 1024
#define FDIM 3584
#define MAXROWS (TTOK * 2)

typedef __attribute__((ext_vector_type(16))) __bf16 v16bf;
typedef __attribute__((ext_vector_type(8)))  __bf16 v8bf;
typedef __attribute__((ext_vector_type(4)))  __bf16 v4bf;
typedef __attribute__((ext_vector_type(8)))  float  v8f;

// LDS row stride (in bf16 elems) for the 16xD staged A tile: 1024 + 8 pad.
// Byte stride 2064 = 129*16 keeps 16B alignment and spreads rows across banks.
#define SA_STRIDE 1032

__device__ __forceinline__ v16bf cat16(v8bf a, v8bf b) {
  return __builtin_shufflevector(a, b, 0,1,2,3,4,5,6,7,8,9,10,11,12,13,14,15);
}

// silu(g) = g * rcp(1 + exp(-g)); hardware v_rcp_f32 instead of IEEE divide.
__device__ __forceinline__ float fast_silu(float g) {
  return g * __builtin_amdgcn_rcpf(1.0f + __expf(-g));
}

// Load 16 K-values of a B fragment for this lane from an fp32 weight row,
// converting to bf16 in-register. k1/k2 are the two 8-wide K chunks this
// lane owns per the CDNA5 16-bit matrix VGPR layout.
__device__ __forceinline__ v16bf load_b_f32(const float* __restrict__ row, int k1, int k2) {
  float4 x0 = *(const float4*)(row + k1);
  float4 x1 = *(const float4*)(row + k1 + 4);
  float4 y0 = *(const float4*)(row + k2);
  float4 y1 = *(const float4*)(row + k2 + 4);
  v16bf b;
  b[0]=(__bf16)x0.x;  b[1]=(__bf16)x0.y;  b[2]=(__bf16)x0.z;  b[3]=(__bf16)x0.w;
  b[4]=(__bf16)x1.x;  b[5]=(__bf16)x1.y;  b[6]=(__bf16)x1.z;  b[7]=(__bf16)x1.w;
  b[8]=(__bf16)y0.x;  b[9]=(__bf16)y0.y;  b[10]=(__bf16)y0.z; b[11]=(__bf16)y0.w;
  b[12]=(__bf16)y1.x; b[13]=(__bf16)y1.y; b[14]=(__bf16)y1.z; b[15]=(__bf16)y1.w;
  return b;
}

// ---------------- Routing / gather (deterministic, tiny) ----------------
__global__ void moe_route_kernel(const int* __restrict__ route,
                                 const float* __restrict__ rw,
                                 int* __restrict__ off,
                                 int* __restrict__ rowTok,
                                 float* __restrict__ rowWt) {
  __shared__ int cnt[NEXP];
  __shared__ int offS[NEXP + 1];
  const int tid = threadIdx.x;
  if (tid < NEXP) {
    int c = 0;
    for (int t = 0; t < TTOK; ++t) {
      int r0 = route[t * 2], r1 = route[t * 2 + 1];
      if (r0 == tid || r1 == tid) c++;
    }
    cnt[tid] = c;
  }
  __syncthreads();
  if (tid == 0) {
    int s = 0;
    for (int e = 0; e < NEXP; ++e) { offS[e] = s; s += cnt[e]; }
    offS[NEXP] = s;
    for (int e = 0; e <= NEXP; ++e) off[e] = offS[e];
  }
  __syncthreads();
  if (tid < NEXP) {
    int pos = offS[tid];
    for (int t = 0; t < TTOK; ++t) {
      int r0 = route[t * 2], r1 = route[t * 2 + 1];
      bool in0 = (r0 == tid), in1 = (r1 == tid);
      if (in0 || in1) {
        float w = (in0 ? rw[t * 2] : 0.0f) + (in1 ? rw[t * 2 + 1] : 0.0f);
        rowTok[pos] = t;
        rowWt[pos]  = w;
        pos++;
      }
    }
  }
}

// ---------------- GEMM1: act = silu(x@w1^T) * (x@w3^T), bf16 out ----------------
// grid = (mtile=128, ntile=F/128=28, expert=8); block = 256 threads (8 waves).
// Each wave owns one 16x16 column group; gate+up accumulated together.
__global__ __launch_bounds__(256) void moe_gemm1_kernel(
    const float* __restrict__ hs, const float* __restrict__ w13,
    const int* __restrict__ off, const int* __restrict__ rowTok,
    __bf16* __restrict__ act) {
  const int e      = blockIdx.z;
  const int mtile  = blockIdx.x;
  const int rowBeg = off[e] + mtile * 16;
  const int rowEnd = off[e + 1];
  if (rowBeg >= rowEnd) return;
  const int colBase = blockIdx.y * 128;  // within F

  __shared__ unsigned short sA[16 * SA_STRIDE];
  __shared__ int sTok[16];

  const int tid = threadIdx.x;
  if (tid < 16) {
    int r = rowBeg + tid;
    sTok[tid] = (r < rowEnd) ? rowTok[r] : -1;
  }
  __syncthreads();

  // Stage 16 x 1024 fp32 rows -> bf16 LDS (zero-fill past rowEnd).
  for (int i = 0; i < 16; ++i) {
    int idx = tid + i * 256;       // 4096 float4 slots total
    int r   = idx >> 8;            // 256 float4 per row
    int c   = (idx & 255) << 2;    // float column
    int t   = sTok[r];
    float4 v = make_float4(0.f, 0.f, 0.f, 0.f);
    if (t >= 0) v = *(const float4*)(hs + (size_t)t * DDIM + c);
    v4bf p = { (__bf16)v.x, (__bf16)v.y, (__bf16)v.z, (__bf16)v.w };
    *(v4bf*)(&sA[r * SA_STRIDE + c]) = p;
  }
  __syncthreads();

  const int lane = tid & 31;
  const int wv   = tid >> 5;
  const int nn   = lane & 15;        // N (and A-row) index for this lane
  const int hv   = lane >> 4;        // half selector
  const int colG = colBase + wv * 16 + nn;   // gate column in [0,F)
  const float* __restrict__ wG = w13 + ((size_t)e * 2 * FDIM + colG) * DDIM;
  const float* __restrict__ wU = w13 + ((size_t)e * 2 * FDIM + FDIM + colG) * DDIM;
  const unsigned short* __restrict__ aRow = &sA[nn * SA_STRIDE];

  v8f accG = {}; v8f accU = {};
  for (int k = 0; k < DDIM; k += 32) {
    const int k1 = k + hv * 8;
    const int k2 = k1 + 16;
    __builtin_prefetch(wG + k1 + 32, 0, 1);
    __builtin_prefetch(wU + k1 + 32, 0, 1);
    v8bf a0 = *(const v8bf*)(aRow + k1);
    v8bf a1 = *(const v8bf*)(aRow + k2);
    v16bf a  = cat16(a0, a1);
    v16bf bG = load_b_f32(wG, k1, k2);
    v16bf bU = load_b_f32(wU, k1, k2);
    accG = __builtin_amdgcn_wmma_f32_16x16x32_bf16(false, a, false, bG, (short)0, accG, false, false);
    accU = __builtin_amdgcn_wmma_f32_16x16x32_bf16(false, a, false, bU, (short)0, accU, false, false);
  }

  // SiLU(gate) * up -> bf16 act. Fast path for full 16-row tiles.
  __bf16* __restrict__ actCol = act + colG;
  if (rowBeg + 16 <= rowEnd) {
#pragma unroll
    for (int r = 0; r < 8; ++r) {
      int m = r + hv * 8;
      actCol[(size_t)(rowBeg + m) * FDIM] = (__bf16)(fast_silu(accG[r]) * accU[r]);
    }
  } else {
#pragma unroll
    for (int r = 0; r < 8; ++r) {
      int m    = r + hv * 8;
      int gRow = rowBeg + m;
      if (gRow < rowEnd) {
        actCol[(size_t)gRow * FDIM] = (__bf16)(fast_silu(accG[r]) * accU[r]);
      }
    }
  }
}

// ---------------- GEMM2: out += rowWt * (act @ w2^T) ----------------
// grid = (mtile=128, ntile=D/128=8, expert=8); block = 256 threads.
__global__ __launch_bounds__(256) void moe_gemm2_kernel(
    const __bf16* __restrict__ act, const float* __restrict__ w2,
    const int* __restrict__ off, const int* __restrict__ rowTok,
    const float* __restrict__ rowWt, float* __restrict__ out) {
  const int e      = blockIdx.z;
  const int mtile  = blockIdx.x;
  const int rowBeg = off[e] + mtile * 16;
  const int rowEnd = off[e + 1];
  if (rowBeg >= rowEnd) return;
  const int colBase = blockIdx.y * 128;  // within D

  const int tid  = threadIdx.x;
  const int lane = tid & 31;
  const int wv   = tid >> 5;
  const int nn   = lane & 15;
  const int hv   = lane >> 4;
  const int col  = colBase + wv * 16 + nn;

  const float* __restrict__ wRow = w2 + ((size_t)e * DDIM + col) * FDIM;
  int aIdx = rowBeg + nn;
  if (aIdx >= rowEnd) aIdx = rowEnd - 1;  // clamp: garbage rows are masked at store
  const __bf16* __restrict__ aRow = act + (size_t)aIdx * FDIM;

  v8f acc = {};
  for (int k = 0; k < FDIM; k += 32) {
    const int k1 = k + hv * 8;
    const int k2 = k1 + 16;
    __builtin_prefetch(wRow + k1 + 32, 0, 1);
    v8bf a0 = *(const v8bf*)(aRow + k1);
    v8bf a1 = *(const v8bf*)(aRow + k2);
    v16bf a = cat16(a0, a1);
    v16bf b = load_b_f32(wRow, k1, k2);
    acc = __builtin_amdgcn_wmma_f32_16x16x32_bf16(false, a, false, b, (short)0, acc, false, false);
  }

  if (rowBeg + 16 <= rowEnd) {
#pragma unroll
    for (int r = 0; r < 8; ++r) {
      int m    = r + hv * 8;
      int gRow = rowBeg + m;
      float y  = acc[r] * rowWt[gRow];
      unsafeAtomicAdd(&out[(size_t)rowTok[gRow] * DDIM + col], y);
    }
  } else {
#pragma unroll
    for (int r = 0; r < 8; ++r) {
      int m    = r + hv * 8;
      int gRow = rowBeg + m;
      if (gRow < rowEnd) {
        float y = acc[r] * rowWt[gRow];
        unsafeAtomicAdd(&out[(size_t)rowTok[gRow] * DDIM + col], y);
      }
    }
  }
}

extern "C" void kernel_launch(void* const* d_in, const int* in_sizes, int n_in,
                              void* d_out, int out_size, void* d_ws, size_t ws_size,
                              hipStream_t stream) {
  const float* hs    = (const float*)d_in[0];
  const int*   route = (const int*)d_in[1];
  const float* rw    = (const float*)d_in[2];
  const float* w13   = (const float*)d_in[3];
  const float* w2    = (const float*)d_in[4];
  float*       out   = (float*)d_out;

  // Workspace layout
  int*    off    = (int*)d_ws;                               // 16 ints (9 used)
  int*    rowTok = off + 16;                                 // MAXROWS
  float*  rowWt  = (float*)(rowTok + MAXROWS);               // MAXROWS
  __bf16* actB   = (__bf16*)((char*)d_ws + 32768);           // MAXROWS x FDIM bf16

  hipMemsetAsync(out, 0, (size_t)TTOK * DDIM * sizeof(float), stream);

  moe_route_kernel<<<1, 32, 0, stream>>>(route, rw, off, rowTok, rowWt);

  dim3 g1(MAXROWS / 16, FDIM / 128, NEXP);
  moe_gemm1_kernel<<<g1, 256, 0, stream>>>(hs, w13, off, rowTok, actB);

  dim3 g2(MAXROWS / 16, DDIM / 128, NEXP);
  moe_gemm2_kernel<<<g2, 256, 0, stream>>>(actB, w2, off, rowTok, rowWt, out);
}
